// SimpleEncoder_33543694582512
// MI455X (gfx1250) — matrix operand order repaired and checked by reading, not compiled, not run on hardware
//
#include <hip/hip_runtime.h>
#include <hip/hip_bf16.h>
#include <math.h>

// ---------------------------------------------------------------------------
// SimpleEncoder for MI455X (gfx1250, wave32, WMMA)
//   Kernel 1: weight prep  (f32 -> f16, pad w1 Cin 10->32)
//   Kernel 2: KNN(32) + relative-feature build, one block per (b,g)
//   Kernel 3: fused MLP chain per group, v_wmma_f32_16x16x32_f16 GEMMs
// ---------------------------------------------------------------------------

typedef __attribute__((ext_vector_type(16))) _Float16 v16h;
typedef __attribute__((ext_vector_type(8)))  float    v8f;

union H16 { v16h v; _Float16 h[16]; uint4 q[2]; };

#define NPTS   8192
#define GROUPS 4096   // B*G = 8*512
#define KNN    32

// ---------------- fragment loaders ------------------------------------------
// A: 16x32 f16 tile, M rows striped over lanes (M = lane&15), K scattered:
//   half h (0..15) -> K = (h>>3)*16 + ((lane>>4)&1)*8 + (h&7)
// -> two contiguous 8-half (16B) runs per lane.
__device__ static inline v16h load_a_frag(const _Float16* lds, int stride,
                                          int mbase, int kbase, int lane) {
    int row = mbase + (lane & 15);
    int hi  = (lane >> 4) & 1;
    const _Float16* p = lds + row * stride + kbase + hi * 8;
    H16 u;
    u.q[0] = *(const uint4*)(p);
    u.q[1] = *(const uint4*)(p + 16);
    return u.v;
}

// B: 32x16 f16 tile from W[Cout][Cin] row-major (B = W^T):
//   N = nbase + (lane&15); half h -> K = kbase + (h>>3)*16 + ((lane>>4)&1)*8 + (h&7)
// -> two contiguous 8-half (16B) runs per lane.
__device__ static inline v16h load_b_frag(const _Float16* W, int Cin,
                                          int nbase, int kbase, int lane) {
    int n  = nbase + (lane & 15);
    int hi = (lane >> 4) & 1;
    const _Float16* p = W + n * Cin + kbase + hi * 8;
    H16 u;
    u.q[0] = *(const uint4*)(p);
    u.q[1] = *(const uint4*)(p + 16);
    return u.v;
}

// ---------------- tiled GEMM: D[32,Ncols] = A[32,Cin] x W^T, post-op --------
// 8 waves (256 threads), wave w owns output tiles t = w, w+8, ...
// relu_bn: out = relu( (acc+bias)*scale + shift ),  scale/shift from BN params
// else:    out = acc + bias
__device__ static inline void gemm_tiles(const _Float16* Alds, int Astride,
                                         const _Float16* Wg,  int Cin,
                                         _Float16* Olds, int Ostride, int Ocol0,
                                         int Ncols, const float* bias,
                                         const float* g, const float* be,
                                         const float* mm, const float* vv,
                                         bool relu_bn) {
    const int tid  = threadIdx.x;
    const int lane = tid & 31;
    const int wave = tid >> 5;
    const int ntot = (Ncols >> 4) * 2;   // n-tiles * 2 m-tiles
    const int kch  = Cin >> 5;
    for (int t = wave; t < ntot; t += 8) {
        const int mt = t & 1;
        const int nt = t >> 1;
        v8f acc = {};
        for (int kc = 0; kc < kch; ++kc) {
            v16h a = load_a_frag(Alds, Astride, mt * 16, kc * 32, lane);
            v16h b = load_b_frag(Wg, Cin, nt * 16, kc * 32, lane);
            acc = __builtin_amdgcn_wmma_f32_16x16x32_f16(
                false, a, false, b, (short)0, acc, false, false);
        }
        const int n  = nt * 16 + (lane & 15);
        const int hi = (lane >> 4) & 1;
        float bb = bias[n];
        float scale = 1.f, shift = 0.f;
        if (relu_bn) {
            float inv = rsqrtf(vv[n] + 1e-5f);
            scale = g[n] * inv;
            shift = be[n] - mm[n] * scale;
        }
        for (int r = 0; r < 8; ++r) {
            int m = mt * 16 + r + hi * 8;
            float v = acc[r] + bb;
            if (relu_bn) { v = v * scale + shift; v = v > 0.f ? v : 0.f; }
            Olds[m * Ostride + Ocol0 + n] = (_Float16)v;
        }
    }
}

// ---------------- attention pool: softmax over 32 rows, then 1xC @ CxC ------
__device__ static inline void attn_pool(const _Float16* X, int stride, int col0,
                                        int C, const float* sw, const float* sbp,
                                        const float* mw, const float* mb,
                                        float* pooled, float* scores,
                                        float* scores2, float* fg_lds,
                                        float* out_g) {
    const int tid = threadIdx.x;
    if (tid < 32) {
        float s = sbp[0];
        for (int c = 0; c < C; ++c)
            s += (float)X[tid * stride + col0 + c] * sw[c];
        scores[tid] = s;
    }
    __syncthreads();
    if (tid < 32) {
        float mx = -3.0e38f;
        for (int r = 0; r < 32; ++r) mx = fmaxf(mx, scores[r]);
        float den = 0.f;
        for (int r = 0; r < 32; ++r) den += __expf(scores[r] - mx);
        scores2[tid] = __expf(scores[tid] - mx) / den;
    }
    __syncthreads();
    for (int c = tid; c < C; c += 256) {
        float acc = 0.f;
        for (int r = 0; r < 32; ++r)
            acc += scores2[r] * (float)X[r * stride + col0 + c];
        pooled[c] = acc;
    }
    __syncthreads();
    for (int o = tid; o < C; o += 256) {
        float acc = mb[o];
        const float* wr = mw + o * C;
        for (int c = 0; c < C; ++c) acc += pooled[c] * wr[c];
        if (fg_lds) fg_lds[o] = acc; else out_g[o] = acc;
    }
    __syncthreads();
}

// ---------------- kernel 1: weight f32->f16 prep ----------------------------
__global__ void wprep_kernel(const float* w1, const float* w2, const float* w3,
                             const float* w4, _Float16* w1h, _Float16* w2h,
                             _Float16* w3h, _Float16* w4h) {
    int i = blockIdx.x * 256 + threadIdx.x;
    if (i < 128 * 32) {  // w1 padded [128][10] -> [128][32]
        int o = i >> 5, c = i & 31;
        w1h[i] = (_Float16)(c < 10 ? w1[o * 10 + c] : 0.f);
    }
    if (i < 256 * 128) w2h[i] = (_Float16)w2[i];
    if (i < 512 * 512) { w3h[i] = (_Float16)w3[i]; w4h[i] = (_Float16)w4[i]; }
}

// ---------------- kernel 2: KNN + feature build -----------------------------
__global__ __launch_bounds__(256)
void knn_feat_kernel(const float* __restrict__ xyz, const int* __restrict__ cidx,
                     _Float16* __restrict__ feat, float* __restrict__ out_center) {
    __shared__ float dist[NPTS];          // 32 KB
    __shared__ float red_v[256];
    __shared__ int   red_i[256];
    __shared__ int   sel[KNN];
    __shared__ float cc[3];

    const int grp = blockIdx.x;           // b*512 + g
    const int b   = grp >> 9;
    const int tid = threadIdx.x;
    const float* xb = xyz + (size_t)b * NPTS * 3;

    if (tid == 0) {
        int ci = cidx[grp];
        cc[0] = xb[ci * 3 + 0];
        cc[1] = xb[ci * 3 + 1];
        cc[2] = xb[ci * 3 + 2];
        out_center[grp * 3 + 0] = cc[0];
        out_center[grp * 3 + 1] = cc[1];
        out_center[grp * 3 + 2] = cc[2];
    }
    __syncthreads();
    const float cx = cc[0], cy = cc[1], cz = cc[2];

    for (int i = tid; i < NPTS; i += 256) {
        float dx = xb[i * 3 + 0] - cx;
        float dy = xb[i * 3 + 1] - cy;
        float dz = xb[i * 3 + 2] - cz;
        dist[i] = dx * dx + dy * dy + dz * dz;
    }
    __syncthreads();

    for (int k = 0; k < KNN; ++k) {
        float bv = 3.4e38f; int bi = 0x7fffffff;
        for (int i = tid; i < NPTS; i += 256) {
            float d = dist[i];
            if (d < bv || (d == bv && i < bi)) { bv = d; bi = i; }
        }
        red_v[tid] = bv; red_i[tid] = bi;
        __syncthreads();
        for (int off = 128; off > 0; off >>= 1) {
            if (tid < off) {
                float ov = red_v[tid + off]; int oi = red_i[tid + off];
                if (ov < red_v[tid] || (ov == red_v[tid] && oi < red_i[tid])) {
                    red_v[tid] = ov; red_i[tid] = oi;
                }
            }
            __syncthreads();
        }
        if (tid == 0) { sel[k] = red_i[0]; dist[red_i[0]] = 3.4e38f; }
        __syncthreads();
    }

    if (tid < KNN) {
        int i = sel[tid];
        float px = xb[i * 3 + 0], py = xb[i * 3 + 1], pz = xb[i * 3 + 2];
        float rx = cx - px, ry = cy - py, rz = cz - pz;
        float rd = sqrtf(rx * rx + ry * ry + rz * rz);
        _Float16* f = feat + ((size_t)grp * KNN + tid) * 32;
        f[0] = (_Float16)rd;
        f[1] = (_Float16)rx; f[2] = (_Float16)ry; f[3] = (_Float16)rz;
        f[4] = (_Float16)cx; f[5] = (_Float16)cy; f[6] = (_Float16)cz;
        f[7] = (_Float16)px; f[8] = (_Float16)py; f[9] = (_Float16)pz;
        for (int c = 10; c < 32; ++c) f[c] = (_Float16)0.f;
    }
}

// ---------------- kernel 3: fused per-group MLP chain -----------------------
__global__ __launch_bounds__(256)
void encoder_mlp_kernel(const _Float16* __restrict__ feat,
                        const _Float16* __restrict__ w1h,
                        const _Float16* __restrict__ w2h,
                        const _Float16* __restrict__ w3h,
                        const _Float16* __restrict__ w4h,
                        const float* b1, const float* g1, const float* be1,
                        const float* m1, const float* v1, const float* b2,
                        const float* ap1_sw, const float* ap1_sb,
                        const float* ap1_mw, const float* ap1_mb,
                        const float* b3, const float* g2, const float* be2,
                        const float* m2, const float* v2, const float* b4,
                        const float* ap2_sw, const float* ap2_sb,
                        const float* ap2_mw, const float* ap2_mb,
                        float* __restrict__ out_fg2) {
    __shared__ __align__(16) _Float16 bufA[32 * 512];  // 32 KB
    __shared__ __align__(16) _Float16 bufB[32 * 512];  // 32 KB
    // scratch overlaid into bufA (cols live only when that region is dead):
    float* pooled  = (float*)(bufA + 16 * 512);        // 2 KB
    float* fgv     = pooled + 512;                     // 2 KB
    float* scores  = fgv + 512;                        // 128 B
    float* scores2 = scores + 32;                      // 128 B

    const int grp = blockIdx.x;
    const int tid = threadIdx.x;

    // load padded features [32][32] f16 into bufB
    for (int i = tid; i < 32 * 32; i += 256)
        bufB[i] = feat[(size_t)grp * 1024 + i];
    __syncthreads();

    // stage 1: X1[32,128] = relu(bn(F @ w1^T + b1)) -> bufA compact stride 128
    gemm_tiles(bufB, 32, w1h, 32, bufA, 128, 0, 128, b1, g1, be1, m1, v1, true);
    __syncthreads();

    // stage 2: X2[32,256] = X1 @ w2^T + b2 -> bufB cols 256..511 (stride 512)
    gemm_tiles(bufA, 128, w2h, 128, bufB, 512, 256, 256, b2, b2, b2, b2, b2, false);
    __syncthreads();

    // attention pool 1 on X2 -> fgv[256] (LDS)
    attn_pool(bufB, 512, 256, 256, ap1_sw, ap1_sb, ap1_mw, ap1_mb,
              pooled, scores, scores2, fgv, nullptr);

    // broadcast fg into X3 cols 0..255 (bufB)
    for (int i = tid; i < 32 * 256; i += 256) {
        int r = i >> 8, c = i & 255;
        bufB[r * 512 + c] = (_Float16)fgv[c];
    }
    __syncthreads();

    // stage 3: X4[32,512] = relu(bn(X3 @ w3^T + b3)) -> bufA (stride 512)
    gemm_tiles(bufB, 512, w3h, 512, bufA, 512, 0, 512, b3, g2, be2, m2, v2, true);
    __syncthreads();

    // stage 4: X5[32,512] = X4 @ w4^T + b4 -> bufB
    gemm_tiles(bufA, 512, w4h, 512, bufB, 512, 0, 512, b4, b4, b4, b4, b4, false);
    __syncthreads();

    // attention pool 2 on X5 -> global output (X4/bufA dead -> scratch reuse ok)
    attn_pool(bufB, 512, 0, 512, ap2_sw, ap2_sb, ap2_mw, ap2_mb,
              pooled, scores, scores2, nullptr, out_fg2 + (size_t)grp * 512);
}

// ---------------------------------------------------------------------------
extern "C" void kernel_launch(void* const* d_in, const int* in_sizes, int n_in,
                              void* d_out, int out_size, void* d_ws, size_t ws_size,
                              hipStream_t stream) {
    const float* xyz    = (const float*)d_in[0];
    const int*   cidx   = (const int*)d_in[1];
    const float* w1     = (const float*)d_in[2];
    const float* b1     = (const float*)d_in[3];
    const float* g1     = (const float*)d_in[4];
    const float* be1    = (const float*)d_in[5];
    const float* m1     = (const float*)d_in[6];
    const float* v1     = (const float*)d_in[7];
    const float* w2     = (const float*)d_in[8];
    const float* b2     = (const float*)d_in[9];
    const float* ap1_sw = (const float*)d_in[10];
    const float* ap1_sb = (const float*)d_in[11];
    const float* ap1_mw = (const float*)d_in[12];
    const float* ap1_mb = (const float*)d_in[13];
    const float* w3     = (const float*)d_in[14];
    const float* b3     = (const float*)d_in[15];
    const float* g2     = (const float*)d_in[16];
    const float* be2    = (const float*)d_in[17];
    const float* m2     = (const float*)d_in[18];
    const float* v2     = (const float*)d_in[19];
    const float* w4     = (const float*)d_in[20];
    const float* b4     = (const float*)d_in[21];
    const float* ap2_sw = (const float*)d_in[22];
    const float* ap2_sb = (const float*)d_in[23];
    const float* ap2_mw = (const float*)d_in[24];
    const float* ap2_mb = (const float*)d_in[25];

    char* ws = (char*)d_ws;
    _Float16* feat = (_Float16*)ws;                                // 8 MB
    _Float16* w1h  = (_Float16*)(ws + (8u << 20));                 // 8 KB
    _Float16* w2h  = (_Float16*)(ws + (8u << 20) + 8192);          // 64 KB
    _Float16* w3h  = (_Float16*)(ws + (8u << 20) + 8192 + 65536);  // 512 KB
    _Float16* w4h  = w3h + 512 * 512;                              // 512 KB

    float* out = (float*)d_out;          // [0,12288): center, then fg2

    wprep_kernel<<<1024, 256, 0, stream>>>(w1, w2, w3, w4, w1h, w2h, w3h, w4h);
    knn_feat_kernel<<<GROUPS, 256, 0, stream>>>(xyz, cidx, feat, out);
    encoder_mlp_kernel<<<GROUPS, 256, 0, stream>>>(
        feat, w1h, w2h, w3h, w4h,
        b1, g1, be1, m1, v1, b2,
        ap1_sw, ap1_sb, ap1_mw, ap1_mb,
        b3, g2, be2, m2, v2, b4,
        ap2_sw, ap2_sb, ap2_mw, ap2_mb,
        out + 8 * 512 * 3);
}
